// GNNModel_15109694947665
// MI455X (gfx1250) — compile-verified, matrix operand order checked by reading, not run on hardware
//
#include <hip/hip_runtime.h>

typedef __bf16 bf16;
typedef __attribute__((ext_vector_type(16))) __bf16 v16bf;
typedef __attribute__((ext_vector_type(8)))  float  v8f;

#define Nn 50000
#define Ee 800000
#define Gg 64
#define Ff 16
#define Hh 128
#define GFd 3

static inline size_t align256(size_t x) { return (x + 255) & ~(size_t)255; }
static inline int blocks_for(long long n) { return (int)((n + 255) / 256); }

// ---------------- utility kernels ----------------

__global__ void k_fill(float* __restrict__ p, long long n, float v) {
    long long t = (long long)blockIdx.x * blockDim.x + threadIdx.x;
    if (t < n) p[t] = v;
}

__global__ void k_deg(float* __restrict__ deg, const int* __restrict__ recv, int e) {
    int t = blockIdx.x * blockDim.x + threadIdx.x;
    if (t < e) atomicAdd(&deg[recv[t]], 1.0f);
}

__global__ void k_rsqrt(float* __restrict__ d, int n) {
    int t = blockIdx.x * blockDim.x + threadIdx.x;
    if (t < n) d[t] = rsqrtf(d[t]);   // deg >= 1 always (self loop)
}

// pad x (N x 16 f32) -> XB (N x 32 bf16, zero-padded K)
__global__ void k_cast_x(const float* __restrict__ x, bf16* __restrict__ xb, long long n) {
    long long t = (long long)blockIdx.x * blockDim.x + threadIdx.x;
    if (t >= n) return;
    int i = (int)(t >> 5), k = (int)(t & 31);
    xb[t] = (k < Ff) ? (bf16)x[(size_t)i * Ff + k] : (bf16)0.0f;
}

// Repack weight W (Ksrc x 128 f32, row-major) into WMMA-fragment-ready bf16:
//   WF[((colTile*(K/32) + kstep)*32 + lane)*16 + j]
// where lane holds column n = colTile*16 + (lane&15) and
//   k = kstep*32 + 8*(lane>>4) + (j&7) + (j>=8 ? 16 : 0)
// Rows k >= Ksrc are zero (K padding for layer 1).
__global__ void k_pack_w(const float* __restrict__ W, bf16* __restrict__ WF,
                         int K, int Ksrc) {
    int t = blockIdx.x * blockDim.x + threadIdx.x;
    int total = (Hh / 16) * (K / 32) * 512;
    if (t >= total) return;
    int j    = t & 15;
    int lane = (t >> 4) & 31;
    int rest = t >> 9;
    int ks   = rest % (K / 32);
    int ct   = rest / (K / 32);
    int n = ct * 16 + (lane & 15);
    int k = ks * 32 + 8 * (lane >> 4) + (j & 7) + ((j >= 8) ? 16 : 0);
    WF[t] = (k < Ksrc) ? (bf16)W[(size_t)k * Hh + n] : (bf16)0.0f;
}

// ---------------- WMMA GEMM: C[N,128] = A[N,K](bf16) x Wfrag(bf16) ----------------
// grid.x = N/16 row tiles, 256 threads = 8 waves, wave w owns 16x16 tile at cols [16w,16w+16)
__global__ void __launch_bounds__(256)
k_gemm_bf16(const bf16* __restrict__ A, const bf16* __restrict__ WF,
            float* __restrict__ C, int K) {
    const int wave = threadIdx.x >> 5;
    const int lane = threadIdx.x & 31;
    const int rowBase = blockIdx.x * 16;
    const int hi  = lane >> 4;
    const int l15 = lane & 15;
    const int arow = rowBase + l15;       // A: lane holds row M = lane&15
    const int col  = wave * 16 + l15;     // C: lane holds column N = lane&15
    const int ksteps = K >> 5;

    v8f c = {};
    for (int ks = 0; ks < ksteps; ++ks) {
        // A fragment (16x32 bf16): lanes 0-15: K[kb..kb+7] ++ K[kb+16..kb+23]; lanes 16-31: +8
        union { v16bf v; uint4 q[2]; } a;
        const bf16* ap = A + (size_t)arow * K + ks * 32 + 8 * hi;
        a.q[0] = *(const uint4*)(ap);
        a.q[1] = *(const uint4*)(ap + 16);
        __builtin_prefetch(ap + 32, 0, 1);

        // B fragment: pre-packed, 16 contiguous bf16 per lane
        union { v16bf v; uint4 q[2]; } b;
        const bf16* bp = WF + ((size_t)(wave * ksteps + ks) * 32 + lane) * 16;
        b.q[0] = *(const uint4*)(bp);
        b.q[1] = *(const uint4*)(bp + 8);

        c = __builtin_amdgcn_wmma_f32_16x16x32_bf16(
                /*neg_a=*/false, a.v, /*neg_b=*/false, b.v,
                /*c_mod=*/(short)0, c, /*reuse_a=*/false, /*reuse_b=*/false);
    }
    // C layout: VGPR v holds M = v (lanes 0-15) / M = v+8 (lanes 16-31)
#pragma unroll
    for (int v = 0; v < 8; ++v)
        C[(size_t)(rowBase + v + 8 * hi) * Hh + col] = c[v];
}

// ---------------- message passing ----------------
// 32 lanes per edge, float4 per lane: AGG[r] += XW[s] * dinv[s]*dinv[r]
__global__ void k_edge_agg(const float* __restrict__ XW, float* __restrict__ AGG,
                           const int* __restrict__ send, const int* __restrict__ recv,
                           const float* __restrict__ dinv, long long n) {
    long long t = (long long)blockIdx.x * blockDim.x + threadIdx.x;
    if (t >= n) return;
    int e = (int)(t >> 5), j = (int)(t & 31);
    int s = send[e], r = recv[e];
    float w = dinv[s] * dinv[r];
    const float4 v = *(const float4*)(XW + (size_t)s * Hh + j * 4);
    float* dst = AGG + (size_t)r * Hh + j * 4;
    atomicAdd(dst + 0, v.x * w);
    atomicAdd(dst + 1, v.y * w);
    atomicAdd(dst + 2, v.z * w);
    atomicAdd(dst + 3, v.w * w);
}

// self loops: exclusive per-element writes, no atomics needed (stream-ordered kernel)
__global__ void k_self_agg(const float* __restrict__ XW, float* __restrict__ AGG,
                           const float* __restrict__ dinv, long long n) {
    long long t = (long long)blockIdx.x * blockDim.x + threadIdx.x;
    if (t >= n) return;
    int i = (int)(t >> 5), j = (int)(t & 31);
    float w = dinv[i] * dinv[i];
    const float4 v = *(const float4*)(XW + (size_t)i * Hh + j * 4);
    float* dst = AGG + (size_t)i * Hh + j * 4;
    dst[0] += v.x * w; dst[1] += v.y * w; dst[2] += v.z * w; dst[3] += v.w * w;
}

// out = relu(AGG + bias), written back f32 (for pooling) and bf16 (next GEMM input)
__global__ void k_bias_relu(float* __restrict__ AGG, const float* __restrict__ bias,
                            bf16* __restrict__ HBF, long long n) {
    long long t = (long long)blockIdx.x * blockDim.x + threadIdx.x;
    if (t >= n) return;
    float v = fmaxf(AGG[t] + bias[(int)(t & (Hh - 1))], 0.0f);
    AGG[t] = v;
    HBF[t] = (bf16)v;
}

// ---------------- pooling + MLP ----------------

__global__ void k_pool_acc(const float* __restrict__ h, const int* __restrict__ batch,
                           float* __restrict__ pooled, float* __restrict__ cnt, long long n) {
    long long t = (long long)blockIdx.x * blockDim.x + threadIdx.x;
    if (t >= n) return;
    int i = (int)(t >> 5), j = (int)(t & 31);
    int g = batch[i];
    const float4 v = *(const float4*)(h + (size_t)i * Hh + j * 4);
    float* dst = pooled + (size_t)g * Hh + j * 4;
    atomicAdd(dst + 0, v.x);
    atomicAdd(dst + 1, v.y);
    atomicAdd(dst + 2, v.z);
    atomicAdd(dst + 3, v.w);
    if (j == 0) atomicAdd(&cnt[g], 1.0f);
}

__global__ void k_build_z(const float* __restrict__ pooled, const float* __restrict__ cnt,
                          const float* __restrict__ gfeat, float* __restrict__ Z) {
    int t = blockIdx.x * blockDim.x + threadIdx.x;
    if (t >= Gg * (Hh + GFd)) return;
    int g = t / (Hh + GFd), d = t % (Hh + GFd);
    Z[t] = (d < Hh) ? pooled[(size_t)g * Hh + d] / fmaxf(cnt[g], 1.0f)
                    : gfeat[(size_t)g * GFd + (d - Hh)];
}

__global__ void k_dense(const float* __restrict__ Z, const float* __restrict__ W,
                        const float* __restrict__ b, float* __restrict__ out,
                        int gdim, int kdim, int odim, int relu) {
    int t = blockIdx.x * blockDim.x + threadIdx.x;
    if (t >= gdim * odim) return;
    int g = t / odim, j = t % odim;
    float acc = b[j];
    for (int k = 0; k < kdim; ++k)
        acc += Z[(size_t)g * kdim + k] * W[(size_t)k * odim + j];
    out[t] = relu ? fmaxf(acc, 0.0f) : acc;
}

// ---------------- launch ----------------

extern "C" void kernel_launch(void* const* d_in, const int* in_sizes, int n_in,
                              void* d_out, int out_size, void* d_ws, size_t ws_size,
                              hipStream_t stream) {
    const float* x      = (const float*)d_in[0];
    const int*   eidx   = (const int*)d_in[1];
    const int*   batch  = (const int*)d_in[2];
    const float* gfeat  = (const float*)d_in[3];
    const float* W1 = (const float*)d_in[4];  const float* b1 = (const float*)d_in[5];
    const float* W2 = (const float*)d_in[6];  const float* b2 = (const float*)d_in[7];
    const float* W3 = (const float*)d_in[8];  const float* b3 = (const float*)d_in[9];
    const float* Wm1 = (const float*)d_in[10]; const float* bm1 = (const float*)d_in[11];
    const float* Wm2 = (const float*)d_in[12]; const float* bm2 = (const float*)d_in[13];
    const float* Wm3 = (const float*)d_in[14]; const float* bm3 = (const float*)d_in[15];
    const int* send = eidx;         // edge_index[0]
    const int* recv = eidx + Ee;    // edge_index[1]
    float* out = (float*)d_out;

    // workspace carve-up
    char* p = (char*)d_ws;
    float* XW   = (float*)p; p += align256(sizeof(float) * (size_t)Nn * Hh);   // 25.6 MB
    float* AGG  = (float*)p; p += align256(sizeof(float) * (size_t)Nn * Hh);   // 25.6 MB
    bf16*  HBF  = (bf16*)p;  p += align256(sizeof(bf16)  * (size_t)Nn * Hh);   // 12.8 MB
    float* dinv = (float*)p; p += align256(sizeof(float) * (size_t)Nn);
    bf16*  WF   = (bf16*)p;  p += align256(sizeof(bf16)  * Hh * Hh);           // packed weights
    float* pooled = (float*)p; p += align256(sizeof(float) * Gg * Hh);
    float* cnt  = (float*)p; p += align256(sizeof(float) * Gg);
    float* Z    = (float*)p; p += align256(sizeof(float) * Gg * (Hh + GFd));
    float* Z1   = (float*)p; p += align256(sizeof(float) * Gg * 64);
    float* Z2   = (float*)p; p += align256(sizeof(float) * Gg * 32);

    const long long NH  = (long long)Nn * Hh;
    const long long E32 = (long long)Ee * 32;
    const long long N32 = (long long)Nn * 32;

    // degree + symmetric norm (self loops counted via init=1)
    k_fill  <<<blocks_for(Nn), 256, 0, stream>>>(dinv, Nn, 1.0f);
    k_deg   <<<blocks_for(Ee), 256, 0, stream>>>(dinv, recv, Ee);
    k_rsqrt <<<blocks_for(Nn), 256, 0, stream>>>(dinv, Nn);

    // ---- layer 1 (K padded 16 -> 32); XB aliases HBF (first 3.2 MB) ----
    k_cast_x   <<<blocks_for(N32), 256, 0, stream>>>(x, HBF, N32);
    k_pack_w   <<<blocks_for(8 * 1 * 512), 256, 0, stream>>>(W1, WF, 32, Ff);
    k_gemm_bf16<<<Nn / 16, 256, 0, stream>>>(HBF, WF, XW, 32);
    k_fill     <<<blocks_for(NH), 256, 0, stream>>>(AGG, NH, 0.0f);
    k_edge_agg <<<blocks_for(E32), 256, 0, stream>>>(XW, AGG, send, recv, dinv, E32);
    k_self_agg <<<blocks_for(N32), 256, 0, stream>>>(XW, AGG, dinv, N32);
    k_bias_relu<<<blocks_for(NH), 256, 0, stream>>>(AGG, b1, HBF, NH);

    // ---- layer 2 ----
    k_pack_w   <<<blocks_for(8 * 4 * 512), 256, 0, stream>>>(W2, WF, Hh, Hh);
    k_gemm_bf16<<<Nn / 16, 256, 0, stream>>>(HBF, WF, XW, Hh);
    k_fill     <<<blocks_for(NH), 256, 0, stream>>>(AGG, NH, 0.0f);
    k_edge_agg <<<blocks_for(E32), 256, 0, stream>>>(XW, AGG, send, recv, dinv, E32);
    k_self_agg <<<blocks_for(N32), 256, 0, stream>>>(XW, AGG, dinv, N32);
    k_bias_relu<<<blocks_for(NH), 256, 0, stream>>>(AGG, b2, HBF, NH);

    // ---- layer 3 ----
    k_pack_w   <<<blocks_for(8 * 4 * 512), 256, 0, stream>>>(W3, WF, Hh, Hh);
    k_gemm_bf16<<<Nn / 16, 256, 0, stream>>>(HBF, WF, XW, Hh);
    k_fill     <<<blocks_for(NH), 256, 0, stream>>>(AGG, NH, 0.0f);
    k_edge_agg <<<blocks_for(E32), 256, 0, stream>>>(XW, AGG, send, recv, dinv, E32);
    k_self_agg <<<blocks_for(N32), 256, 0, stream>>>(XW, AGG, dinv, N32);
    k_bias_relu<<<blocks_for(NH), 256, 0, stream>>>(AGG, b3, HBF, NH);   // AGG = final h (f32)

    // ---- global mean pool + head MLP ----
    k_fill    <<<blocks_for(Gg * Hh), 256, 0, stream>>>(pooled, Gg * Hh, 0.0f);
    k_fill    <<<blocks_for(Gg), 256, 0, stream>>>(cnt, Gg, 0.0f);
    k_pool_acc<<<blocks_for(N32), 256, 0, stream>>>(AGG, batch, pooled, cnt, N32);
    k_build_z <<<blocks_for(Gg * (Hh + GFd)), 256, 0, stream>>>(pooled, cnt, gfeat, Z);
    k_dense   <<<blocks_for(Gg * 64), 256, 0, stream>>>(Z,  Wm1, bm1, Z1, Gg, Hh + GFd, 64, 1);
    k_dense   <<<blocks_for(Gg * 32), 256, 0, stream>>>(Z1, Wm2, bm2, Z2, Gg, 64, 32, 1);
    k_dense   <<<blocks_for(Gg * 1), 256, 0, stream>>>(Z2, Wm3, bm3, out, Gg, 32, 1, 0);
}
